// HumanAware_53283364274191
// MI455X (gfx1250) — matrix operand order repaired and checked by reading, not compiled, not run on hardware
//
#include <hip/hip_runtime.h>
#include <hip/hip_bf16.h>

typedef _Float16 half_t;
typedef __attribute__((ext_vector_type(16))) _Float16 v16h;
typedef __attribute__((ext_vector_type(8)))  float    v8f;

// ---------------------------------------------------------------------------
// Pack fp32 conv weights into per-lane WMMA A-fragment order:
//   Afrag[((mt*nK + ks)*32 + lane)*16 + e]
// where for lane (hs = lane>>4, m = mt*16 + (lane&15)):
//   e in [0,16): k = ks*32 + hs*16 + e
// K order is tap-major: k = (ky*KW+kx)*Cin + ci.  Zero-padded.
// transposed==1 : weight tensor is (Cin, Cout, KH, KW)  (ConvTranspose2d)
// ---------------------------------------------------------------------------
__global__ void pack_weights_k(const float* __restrict__ w, half_t* __restrict__ Afrag,
                               int Cout, int Cin, int KHKW, int nMt, int nK,
                               int transposed)
{
    int idx = blockIdx.x * blockDim.x + threadIdx.x;
    int total = nMt * nK * 512;
    if (idx >= total) return;
    int e    = idx & 15;
    int lane = (idx >> 4) & 31;
    int ks   = (idx >> 9) % nK;
    int mt   = (idx >> 9) / nK;
    int m    = mt * 16 + (lane & 15);
    int hs   = lane >> 4;
    int k    = ks * 32 + hs * 16 + e;
    float v = 0.f;
    if (m < Cout && k < Cin * KHKW) {
        int t  = k / Cin;
        int ci = k - t * Cin;
        v = transposed ? w[((size_t)ci * Cout + m) * KHKW + t]
                       : w[((size_t)m * Cin + ci) * KHKW + t];
    }
    Afrag[idx] = (half_t)v;
}

// ---------------------------------------------------------------------------
// Implicit-GEMM conv / transposed-conv, v_wmma_f32_16x16x32_f16.
// One wave: NT x 16 output positions x all Cout (MT 16-row tiles = Cout/16).
// Fast path (Cin%32==0): one tap + one source per K-step; clamped address +
// zero-gate B gather (no per-element predication); A fragments are single
// v16h loads in exact register layout; each A feeds NT WMMAs.
// attMode: 0 none, 1 x*att, 2 x*(1-att)     act: 0 none, 1 relu, 2 sigmoid
// ---------------------------------------------------------------------------
template<int KH, int KW, int STRIDE, int PAD, int TRANSPOSED, int MT, int NT>
__launch_bounds__(256)
__global__ void conv_wmma_k(
    const float* __restrict__ in0, int C0,
    const float* __restrict__ in1, int C1,
    const float* __restrict__ in2, int C2,
    const float* __restrict__ att, int attMode,
    const half_t* __restrict__ Apack,
    const float* __restrict__ bias,
    float* __restrict__ out,
    int N, int H, int W, int Cout, int OH, int OW,
    int Ktot, int Kceil, int act)
{
    constexpr int KHKW = KH * KW;
    const int lane   = threadIdx.x & 31;
    const int wave   = threadIdx.x >> 5;
    const int Npos   = N * OH * OW;
    const int ntiles = (Npos + 15) >> 4;
    const int wtiles = (ntiles + NT - 1) / NT;
    const int wt     = blockIdx.x * 8 + wave;        // wave-uniform
    if (wt >= wtiles) return;

    const int nK      = Kceil >> 5;
    const int halfSel = lane >> 4;
    const int nlane   = lane & 15;
    const int Cin     = C0 + C1 + C2;
    const size_t HW   = (size_t)H * W;

    int n_[NT], oy_[NT], ox_[NT]; bool pv[NT];
#pragma unroll
    for (int j = 0; j < NT; ++j) {
        const int pos = (wt * NT + j) * 16 + nlane;
        pv[j] = (pos < Npos);
        const int pc = pv[j] ? pos : 0;
        n_[j]  = pc / (OH * OW);
        const int pr = pc % (OH * OW);
        oy_[j] = pr / OW;
        ox_[j] = pr % OW;
    }

    v8f acc[MT * NT];
#pragma unroll
    for (int i = 0; i < MT * NT; ++i) acc[i] = (v8f){};

    const half_t* __restrict__ afrag = Apack + (size_t)lane * 16;

    if ((Cin & 31) == 0) {
        // ---------------- fast path: Kceil == Ktot, tap-major ----------------
#pragma unroll 1
        for (int t = 0; t < KHKW; ++t) {
            const int ky = t / KW, kx = t % KW;      // compile-time divisors
            float gatef[NT]; size_t sp[NT];
#pragma unroll
            for (int j = 0; j < NT; ++j) {
                int iy, ix; bool ok;
                if (!TRANSPOSED) {
                    iy = oy_[j] * STRIDE - PAD + ky;
                    ix = ox_[j] * STRIDE - PAD + kx;
                    ok = (iy >= 0) & (iy < H) & (ix >= 0) & (ix < W);
                } else {
                    const int ty = oy_[j] + PAD - ky;
                    const int tx = ox_[j] + PAD - kx;
                    ok = (ty >= 0) & (tx >= 0) &
                         ((ty % STRIDE) == 0) & ((tx % STRIDE) == 0);
                    iy = ty / STRIDE; ix = tx / STRIDE;
                    ok = ok & (iy < H) & (ix < W);
                }
                ok &= pv[j];
                const int iyc = min(max(iy, 0), H - 1);
                const int ixc = min(max(ix, 0), W - 1);
                sp[j] = (size_t)iyc * W + ixc;
                float g = 1.f;
                if (attMode && ok) {
                    const float a = att[(size_t)n_[j] * HW + sp[j]];
                    g = (attMode == 1) ? a : (1.f - a);
                }
                gatef[j] = ok ? g : 0.f;
            }
#pragma unroll 1
            for (int ci0 = 0; ci0 < Cin; ci0 += 32) {
                // one source per 32-aligned channel block (all Cx % 32 == 0)
                const float* __restrict__ src; int cl, Cs;
                if (ci0 < C0)           { src = in0; cl = ci0;           Cs = C0; }
                else if (ci0 < C0 + C1) { src = in1; cl = ci0 - C0;      Cs = C1; }
                else                    { src = in2; cl = ci0 - C0 - C1; Cs = C2; }
                v16h B[NT];
#pragma unroll
                for (int j = 0; j < NT; ++j) {
                    const float* __restrict__ p =
                        src + ((size_t)n_[j] * Cs + cl + halfSel * 16) * HW + sp[j];
#pragma unroll
                    for (int h = 0; h < 16; ++h)
                        B[j][h] = (half_t)(p[(size_t)h * HW] * gatef[j]);
                }
                const int ks = (t * Cin + ci0) >> 5;
#pragma unroll
                for (int mt = 0; mt < MT; ++mt) {
                    const v16h A = *(const v16h*)(afrag + ((size_t)(mt * nK + ks) << 9));
#pragma unroll
                    for (int j = 0; j < NT; ++j)
                        acc[mt * NT + j] = __builtin_amdgcn_wmma_f32_16x16x32_f16(
                            false, A, false, B[j], (short)0, acc[mt * NT + j],
                            false, false);
                }
            }
        }
    } else {
        // -------- generic path (tiny Cin==3 layers): per-element decode ------
#pragma unroll 1
        for (int k0 = 0; k0 < Kceil; k0 += 32) {
            const int kb = k0 + halfSel * 16;
            v16h B[NT];
#pragma unroll
            for (int j = 0; j < NT; ++j) {
#pragma unroll
                for (int h = 0; h < 16; ++h) {
                    const int k = kb + h;
                    float v = 0.f;
                    if (pv[j] && k < Ktot) {
                        const int t  = (unsigned)k / (unsigned)Cin;
                        const int ci = k - t * Cin;
                        const int ky = t / KW, kx = t % KW;
                        int iy, ix; bool ok;
                        if (!TRANSPOSED) {
                            iy = oy_[j] * STRIDE - PAD + ky;
                            ix = ox_[j] * STRIDE - PAD + kx;
                            ok = (iy >= 0) & (iy < H) & (ix >= 0) & (ix < W);
                        } else {
                            const int ty = oy_[j] + PAD - ky;
                            const int tx = ox_[j] + PAD - kx;
                            ok = (ty >= 0) & (tx >= 0) &
                                 ((ty % STRIDE) == 0) & ((tx % STRIDE) == 0);
                            iy = ty / STRIDE; ix = tx / STRIDE;
                            ok = ok & (iy < H) & (ix < W);
                        }
                        if (ok) {
                            const float* __restrict__ src; int c, Cs;
                            if (ci < C0)           { src = in0; c = ci;           Cs = C0; }
                            else if (ci < C0 + C1) { src = in1; c = ci - C0;      Cs = C1; }
                            else                   { src = in2; c = ci - C0 - C1; Cs = C2; }
                            v = src[((size_t)n_[j] * Cs + c) * HW + (size_t)iy * W + ix];
                            if (attMode) {
                                const float a =
                                    att[(size_t)n_[j] * HW + (size_t)iy * W + ix];
                                v *= (attMode == 1) ? a : (1.f - a);
                            }
                        }
                    }
                    B[j][h] = (half_t)v;
                }
            }
            const int ks = k0 >> 5;
#pragma unroll
            for (int mt = 0; mt < MT; ++mt) {
                const v16h A = *(const v16h*)(afrag + ((size_t)(mt * nK + ks) << 9));
#pragma unroll
                for (int j = 0; j < NT; ++j)
                    acc[mt * NT + j] = __builtin_amdgcn_wmma_f32_16x16x32_f16(
                        false, A, false, B[j], (short)0, acc[mt * NT + j],
                        false, false);
            }
        }
    }

    // ---- epilogue: bias + activation + store (32-bit 16x16 C/D layout) ----
#pragma unroll
    for (int j = 0; j < NT; ++j) {
        if (pv[j]) {
            float* __restrict__ op =
                out + (size_t)n_[j] * Cout * OH * OW + (size_t)oy_[j] * OW + ox_[j];
#pragma unroll
            for (int mt = 0; mt < MT; ++mt) {
#pragma unroll
                for (int r = 0; r < 8; ++r) {
                    const int co = mt * 16 + halfSel * 8 + r;
                    if (co < Cout) {
                        float v = acc[mt * NT + j][r] + bias[co];
                        if (act == 1)      v = fmaxf(v, 0.f);
                        else if (act == 2) v = 1.f / (1.f + __expf(-v));
                        op[(size_t)co * OH * OW] = v;
                    }
                }
            }
        }
    }
}

// ---------------------------------------------------------------------------
// Host-side driver
// ---------------------------------------------------------------------------
template<int KH, int KW, int S, int P, int T, int MT, int NT>
static void launch_conv(hipStream_t stream,
                        const float* in0, int C0, const float* in1, int C1,
                        const float* in2, int C2,
                        const float* att, int attMode,
                        const half_t* A, const float* bias, float* out,
                        int Nb, int H, int W, int Cout, int OH, int OW, int act)
{
    const int Cin   = C0 + C1 + C2;
    const int Ktot  = Cin * KH * KW;
    const int Kceil = (Ktot + 31) & ~31;
    const long Npos = (long)Nb * OH * OW;
    const int ntiles = (int)((Npos + 15) >> 4);
    const int wtiles = (ntiles + NT - 1) / NT;
    const int blocks = (wtiles + 7) / 8;
    conv_wmma_k<KH, KW, S, P, T, MT, NT><<<blocks, 256, 0, stream>>>(
        in0, C0, in1, C1, in2, C2, att, attMode, A, bias, out,
        Nb, H, W, Cout, OH, OW, Ktot, Kceil, act);
}

extern "C" void kernel_launch(void* const* d_in, const int* in_sizes, int n_in,
                              void* d_out, int out_size, void* d_ws, size_t ws_size,
                              hipStream_t stream)
{
    (void)in_sizes; (void)n_in; (void)out_size; (void)ws_size;

    // inputs: img first, then params in sorted-key order
    const float* img  = (const float*)d_in[0];
    const float* a1_b = (const float*)d_in[1];
    const float* a1_w = (const float*)d_in[2];
    const float* a2_b = (const float*)d_in[3];
    const float* a2_w = (const float*)d_in[4];
    const float* b1_b = (const float*)d_in[5];
    const float* b1_w = (const float*)d_in[6];
    const float* b2_b = (const float*)d_in[7];
    const float* b2_w = (const float*)d_in[8];
    const float* b3_b = (const float*)d_in[9];
    const float* b3_w = (const float*)d_in[10];
    const float* bo_b = (const float*)d_in[11];
    const float* bo_w = (const float*)d_in[12];
    const float* ds_b = (const float*)d_in[13];
    const float* ds_w = (const float*)d_in[14];
    const float* e1_b = (const float*)d_in[15];
    const float* e1_w = (const float*)d_in[16];
    const float* e2_b = (const float*)d_in[17];
    const float* e2_w = (const float*)d_in[18];
    const float* e3_b = (const float*)d_in[19];
    const float* e3_w = (const float*)d_in[20];
    const float* f1_b = (const float*)d_in[21];
    const float* f1_w = (const float*)d_in[22];
    const float* f2_b = (const float*)d_in[23];
    const float* f2_w = (const float*)d_in[24];
    const float* f3_b = (const float*)d_in[25];
    const float* f3_w = (const float*)d_in[26];
    const float* fo_b = (const float*)d_in[27];
    const float* fo_w = (const float*)d_in[28];
    const float* p1_b = (const float*)d_in[29];
    const float* p1_w = (const float*)d_in[30];
    const float* p2_b = (const float*)d_in[31];
    const float* p2_w = (const float*)d_in[32];
    const float* p2m_b = (const float*)d_in[33];
    const float* p2m_w = (const float*)d_in[34];
    const float* p3_b = (const float*)d_in[35];
    const float* p3_w = (const float*)d_in[36];
    const float* p3m_b = (const float*)d_in[37];
    const float* p3m_w = (const float*)d_in[38];
    const float* po_b = (const float*)d_in[39];
    const float* po_w = (const float*)d_in[40];

    const int Nb = 8;
    const size_t IMGC = (size_t)Nb * 3 * 256 * 256;

    float* out_p   = (float*)d_out;
    float* out_fg  = out_p  + IMGC;
    float* out_bg  = out_fg + IMGC;
    float* out_att = out_bg + IMGC;

    char* ws = (char*)d_ws;
    size_t off = 0;
    auto alloc = [&](size_t bytes) -> void* {
        void* p = ws + off;
        off = (off + bytes + 255) & ~(size_t)255;
        return p;
    };

    auto pack = [&](const float* w, int Cout, int Cin, int KH, int KW,
                    int transposed) -> half_t* {
        const int Ktot  = Cin * KH * KW;
        const int Kceil = (Ktot + 31) & ~31;
        const int nK    = Kceil >> 5;
        const int nMt   = ((Cout + 15) & ~15) >> 4;
        half_t* A = (half_t*)alloc((size_t)nMt * nK * 512 * sizeof(half_t));
        const int total = nMt * nK * 512;
        pack_weights_k<<<(total + 255) / 256, 256, 0, stream>>>(
            w, A, Cout, Cin, KH * KW, nMt, nK, transposed);
        return A;
    };

    // ---- pack all weights (f16, per-lane WMMA fragment layout) ----
    half_t* A_ds  = pack(ds_w,  3,   3, 5, 5, 0);
    half_t* A_a1  = pack(a1_w, 32,   3, 3, 3, 0);
    half_t* A_a2  = pack(a2_w,  1,  32, 3, 3, 0);
    half_t* A_e1  = pack(e1_w, 32,   3, 3, 3, 0);
    half_t* A_e2  = pack(e2_w, 64,  32, 3, 3, 0);
    half_t* A_e3  = pack(e3_w, 128, 64, 3, 3, 0);
    half_t* A_f1  = pack(f1_w, 128, 128, 3, 3, 0);
    half_t* A_f2  = pack(f2_w, 64,  128, 4, 4, 1);
    half_t* A_f3  = pack(f3_w, 32,   64, 4, 4, 1);
    half_t* A_fo  = pack(fo_w,  3,   32, 3, 3, 0);
    half_t* A_b1  = pack(b1_w, 128, 128, 3, 3, 0);
    half_t* A_b2  = pack(b2_w, 64,  128, 4, 4, 1);
    half_t* A_b3  = pack(b3_w, 32,   64, 4, 4, 1);
    half_t* A_bo  = pack(bo_w,  3,   32, 3, 3, 0);
    half_t* A_p1  = pack(p1_w, 128, 384, 3, 3, 0);
    half_t* A_p2  = pack(p2_w, 64,  128, 4, 4, 1);
    half_t* A_p2m = pack(p2m_w, 64, 192, 3, 3, 0);
    half_t* A_p3  = pack(p3_w, 32,   64, 4, 4, 1);
    half_t* A_p3m = pack(p3m_w, 32,  96, 3, 3, 0);
    half_t* A_po  = pack(po_w,  3,   32, 3, 3, 0);

    // ---- activation buffers (with reuse) ----
    float* dsb   = (float*)alloc((size_t)Nb * 3 * 64 * 64 * 4);
    float* bigA  = (float*)alloc((size_t)Nb * 32 * 256 * 256 * 4);   // a1h / e1 / p3h
    float* a1hs  = (float*)alloc((size_t)Nb * 32 * 64 * 64 * 4);
    float* attfg = (float*)alloc((size_t)Nb * 1 * 64 * 64 * 4);
    float* e2b   = (float*)alloc((size_t)Nb * 64 * 128 * 128 * 4);   // e2 / p2h
    float* prim  = (float*)alloc((size_t)Nb * 128 * 64 * 64 * 4);
    float* p1h   = (float*)alloc((size_t)Nb * 128 * 64 * 64 * 4);
    float* fl1   = (float*)alloc((size_t)Nb * 128 * 64 * 64 * 4);
    float* fl2   = (float*)alloc((size_t)Nb * 64 * 128 * 128 * 4);
    float* fl3   = (float*)alloc((size_t)Nb * 32 * 256 * 256 * 4);
    float* bl1   = (float*)alloc((size_t)Nb * 128 * 64 * 64 * 4);
    float* bl2   = (float*)alloc((size_t)Nb * 64 * 128 * 128 * 4);
    float* bl3   = (float*)alloc((size_t)Nb * 32 * 256 * 256 * 4);
    float* p2mh  = (float*)alloc((size_t)Nb * 64 * 128 * 128 * 4);
    float* p3mh  = (float*)alloc((size_t)Nb * 32 * 256 * 256 * 4);

    // ---- network ----
    // attention on full image -> att_orig
    launch_conv<3,3,1,1,0,2,1>(stream, img,3, 0,0, 0,0, 0,0, A_a1, a1_b, bigA,    Nb,256,256,32,256,256, 1);
    launch_conv<3,3,1,1,0,1,1>(stream, bigA,32, 0,0, 0,0, 0,0, A_a2, a2_b, out_att,Nb,256,256, 1,256,256, 2);
    // downsample + attention -> att_fg
    launch_conv<5,5,4,2,0,1,1>(stream, img,3, 0,0, 0,0, 0,0, A_ds, ds_b, dsb,     Nb,256,256, 3, 64, 64, 0);
    launch_conv<3,3,1,1,0,2,1>(stream, dsb,3, 0,0, 0,0, 0,0, A_a1, a1_b, a1hs,    Nb, 64, 64,32, 64, 64, 1);
    launch_conv<3,3,1,1,0,1,1>(stream, a1hs,32, 0,0, 0,0, 0,0, A_a2, a2_b, attfg, Nb, 64, 64, 1, 64, 64, 2);
    // encoder
    launch_conv<3,3,1,1,0,2,1>(stream, img,3, 0,0, 0,0, 0,0, A_e1, e1_b, bigA,    Nb,256,256,32,256,256, 1);
    launch_conv<3,3,2,1,0,4,2>(stream, bigA,32, 0,0, 0,0, 0,0, A_e2, e2_b, e2b,   Nb,256,256,64,128,128, 1);
    launch_conv<3,3,2,1,0,8,2>(stream, e2b,64, 0,0, 0,0, 0,0, A_e3, e3_b, prim,   Nb,128,128,128,64, 64, 1);
    // FG decoder (att gating fused into f1 gather)
    launch_conv<3,3,1,1,0,8,2>(stream, prim,128, 0,0, 0,0, attfg,1, A_f1, f1_b, fl1, Nb,64,64,128,64,64, 1);
    launch_conv<4,4,2,1,1,4,2>(stream, fl1,128, 0,0, 0,0, 0,0, A_f2, f2_b, fl2,   Nb, 64, 64,64,128,128, 1);
    launch_conv<4,4,2,1,1,2,1>(stream, fl2,64, 0,0, 0,0, 0,0, A_f3, f3_b, fl3,    Nb,128,128,32,256,256, 1);
    launch_conv<3,3,1,1,0,1,1>(stream, fl3,32, 0,0, 0,0, 0,0, A_fo, fo_b, out_fg, Nb,256,256, 3,256,256, 0);
    // BG decoder (1 - att gating)
    launch_conv<3,3,1,1,0,8,2>(stream, prim,128, 0,0, 0,0, attfg,2, A_b1, b1_b, bl1, Nb,64,64,128,64,64, 1);
    launch_conv<4,4,2,1,1,4,2>(stream, bl1,128, 0,0, 0,0, 0,0, A_b2, b2_b, bl2,   Nb, 64, 64,64,128,128, 1);
    launch_conv<4,4,2,1,1,2,1>(stream, bl2,64, 0,0, 0,0, 0,0, A_b3, b3_b, bl3,    Nb,128,128,32,256,256, 1);
    launch_conv<3,3,1,1,0,1,1>(stream, bl3,32, 0,0, 0,0, 0,0, A_bo, bo_b, out_bg, Nb,256,256, 3,256,256, 0);
    // primary decoder (virtual concats via multi-source gather)
    launch_conv<3,3,1,1,0,8,2>(stream, prim,128, fl1,128, bl1,128, 0,0, A_p1, p1_b, p1h, Nb,64,64,128,64,64, 1);
    launch_conv<4,4,2,1,1,4,2>(stream, p1h,128, 0,0, 0,0, 0,0, A_p2, p2_b, e2b /*p2h*/, Nb,64,64,64,128,128, 1);
    launch_conv<3,3,1,1,0,4,2>(stream, e2b,64, fl2,64, bl2,64, 0,0, A_p2m, p2m_b, p2mh, Nb,128,128,64,128,128, 1);
    launch_conv<4,4,2,1,1,2,1>(stream, p2mh,64, 0,0, 0,0, 0,0, A_p3, p3_b, bigA /*p3h*/, Nb,128,128,32,256,256, 1);
    launch_conv<3,3,1,1,0,2,1>(stream, bigA,32, fl3,32, bl3,32, 0,0, A_p3m, p3m_b, p3mh, Nb,256,256,32,256,256, 1);
    launch_conv<3,3,1,1,0,1,1>(stream, p3mh,32, 0,0, 0,0, 0,0, A_po, po_b, out_p, Nb,256,256, 3,256,256, 0);
}
